// Mamba2Block_8538394985201
// MI455X (gfx1250) — matrix operand order repaired and checked by reading, not compiled, not run on hardware
//
#include <hip/hip_runtime.h>
#include <hip/hip_bf16.h>
#include <cstddef>
#include <cstdint>

// ---------------------------------------------------------------------------
// Mamba2 block for MI455X (gfx1250, wave32, WMMA).
// GEMMs: bf16 WMMA (v_wmma_f32_16x16x32_bf16), f32 accumulate,
// double-buffered LDS fed by GLOBAL_LOAD_ASYNC_TO_LDS_B128 when available.
// ---------------------------------------------------------------------------

typedef __attribute__((ext_vector_type(16))) __bf16 v16bf;
typedef __attribute__((ext_vector_type(8)))  float  v8f;
typedef int v4i_vs __attribute__((vector_size(16)));   // matches builtin's pointee type

#define DIM      1024
#define D_INNER  2048
#define D_STATE  64
#define N_HEADS  32
#define IN_FEAT  4256   // 2*D_INNER + 2*D_STATE + N_HEADS
#define BT       2048   // B*T rows

// workspace layout (bytes)
static constexpr size_t OFF_XBF  = 0;                                // 2048*1024 bf16
static constexpr size_t OFF_WIBF = OFF_XBF  + (size_t)BT*DIM*2;      // 4256*1024 bf16
static constexpr size_t OFF_WOBF = OFF_WIBF + (size_t)IN_FEAT*DIM*2; // 1024*2048 bf16
static constexpr size_t OFF_PROJ = OFF_WOBF + (size_t)DIM*D_INNER*2; // 2048*4256 f32
static constexpr size_t OFF_XSSM = OFF_PROJ + (size_t)BT*IN_FEAT*4;  // 2048*2048 f32
static constexpr size_t OFF_ABAR = OFF_XSSM + (size_t)BT*D_INNER*4;  // 2048*32 f32
static constexpr size_t OFF_Y    = OFF_ABAR + (size_t)BT*N_HEADS*4;  // 2048*2048 f32
static constexpr size_t OFF_YNBF = OFF_Y    + (size_t)BT*D_INNER*4;  // 2048*2048 bf16

// ---------------------------------------------------------------------------
// Async global->LDS copy (CDNA5 ASYNCcnt path) with sync fallback
// ---------------------------------------------------------------------------
#if defined(__HIP_DEVICE_COMPILE__) && __has_builtin(__builtin_amdgcn_global_load_async_to_lds_b128)
#define HAS_ASYNC_LDS 1
#else
#define HAS_ASYNC_LDS 0
#endif

__device__ __forceinline__ void copy_b128_to_lds(const void* g, void* l) {
#if HAS_ASYNC_LDS
    __builtin_amdgcn_global_load_async_to_lds_b128(
        (__attribute__((address_space(1))) v4i_vs*)(uintptr_t)g,
        (__attribute__((address_space(3))) v4i_vs*)(unsigned)(uintptr_t)l,
        0, 0);
#else
    *(uint4*)l = *(const uint4*)g;
#endif
}

__device__ __forceinline__ void wait_async_copies() {
#if HAS_ASYNC_LDS
#if __has_builtin(__builtin_amdgcn_s_wait_asynccnt)
    __builtin_amdgcn_s_wait_asynccnt(0);
#else
    asm volatile("s_wait_asynccnt 0x0" ::: "memory");
#endif
#endif
}

// ---------------------------------------------------------------------------
// f32 -> bf16 cast
// ---------------------------------------------------------------------------
__global__ void cast_f32_bf16(const float* __restrict__ src,
                              __bf16* __restrict__ dst, int n) {
    int i = blockIdx.x * 256 + threadIdx.x;
    if (i < n) dst[i] = (__bf16)src[i];
}

// ---------------------------------------------------------------------------
// bf16 WMMA GEMM: C[m,n] = sum_k A[m,k] * Bw[n,k]   (both K-contiguous)
// WG: 256 thr = 8 waves (2 along M x 4 along N); WG tile 128x256, KT=32.
// Per wave: 4 M-tiles x 4 N-tiles of 16x16 = 16 f32 accumulators.
// Double-buffered LDS staged by async b128 copies.
// ---------------------------------------------------------------------------
#define BM 128
#define BN 256
#define BK 32

union FragAB { v16bf v; uint4 q[2]; };

__global__ __launch_bounds__(256)
void gemm_bf16_wmma(const __bf16* __restrict__ A, const __bf16* __restrict__ Bw,
                    float* __restrict__ C, int M, int N, int K) {
    __shared__ __bf16 As[2][BM * BK];   // 2 x 8 KB
    __shared__ __bf16 Bs[2][BN * BK];   // 2 x 16 KB

    const int tid   = threadIdx.x;
    const int lane  = tid & 31;
    const int wave  = tid >> 5;
    const int waveM = wave & 1;     // 0..1  -> 64 rows each
    const int waveN = wave >> 1;    // 0..3  -> 64 cols each
    const int m0    = blockIdx.y * BM;
    const int n0    = blockIdx.x * BN;
    const int lm    = lane & 15;    // row/col within 16x16 tile
    const int lh    = lane >> 4;    // K-half selector

    // staging: each thread copies 2 A-chunks + 4 B-chunks of 16 bytes per tile
    const int rr = tid >> 2;            // 0..63
    const int sg = (tid & 3) * 8;       // element offset within the 32-wide K slab
    const __bf16* gA[2];
    const __bf16* gB[4];
    int lofsA[2], lofsB[4];
    #pragma unroll
    for (int j = 0; j < 2; ++j) {
        gA[j]    = A + (size_t)min(m0 + rr + 64 * j, M - 1) * K + sg;
        lofsA[j] = (rr + 64 * j) * BK + sg;
    }
    #pragma unroll
    for (int j = 0; j < 4; ++j) {
        gB[j]    = Bw + (size_t)min(n0 + rr + 64 * j, N - 1) * K + sg;
        lofsB[j] = (rr + 64 * j) * BK + sg;
    }

    v8f acc[4][4];
    v8f zero = {};
    #pragma unroll
    for (int i = 0; i < 4; ++i)
        #pragma unroll
        for (int j = 0; j < 4; ++j) acc[i][j] = zero;

    const int nk = K / BK;

    // prologue: stage tile 0 into buffer 0
    #pragma unroll
    for (int j = 0; j < 2; ++j) copy_b128_to_lds(gA[j], &As[0][lofsA[j]]);
    #pragma unroll
    for (int j = 0; j < 4; ++j) copy_b128_to_lds(gB[j], &Bs[0][lofsB[j]]);
    wait_async_copies();
    __syncthreads();

    for (int kt = 0; kt < nk; ++kt) {
        const int p = kt & 1;
        if (kt + 1 < nk) {          // stage next tile into the other buffer
            const int ko = (kt + 1) * BK;
            #pragma unroll
            for (int j = 0; j < 2; ++j) copy_b128_to_lds(gA[j] + ko, &As[p ^ 1][lofsA[j]]);
            #pragma unroll
            for (int j = 0; j < 4; ++j) copy_b128_to_lds(gB[j] + ko, &Bs[p ^ 1][lofsB[j]]);
        }

        // A fragments: lane holds K in {8h..8h+7} U {16+8h..16+8h+7} of its row
        FragAB af[4];
        #pragma unroll
        for (int mt = 0; mt < 4; ++mt) {
            int mrow = waveM * 64 + mt * 16 + lm;
            af[mt].q[0] = *(const uint4*)&As[p][mrow * BK + lh * 8];
            af[mt].q[1] = *(const uint4*)&As[p][mrow * BK + 16 + lh * 8];
        }
        #pragma unroll
        for (int nt = 0; nt < 4; ++nt) {
            // B fragment: lane holds K in {8h..8h+7} U {16+8h..16+8h+7} of W row
            FragAB bfr;
            int nrow = waveN * 64 + nt * 16 + lm;
            bfr.q[0] = *(const uint4*)&Bs[p][nrow * BK + lh * 8];
            bfr.q[1] = *(const uint4*)&Bs[p][nrow * BK + 16 + lh * 8];
            #pragma unroll
            for (int mt = 0; mt < 4; ++mt) {
                acc[mt][nt] = __builtin_amdgcn_wmma_f32_16x16x32_bf16(
                    false, af[mt].v, false, bfr.v,
                    (short)0, acc[mt][nt], false, false);
            }
        }

        wait_async_copies();
        __syncthreads();
    }

    // D layout: lane n = lm; VGPR v -> row v + 8*lh
    for (int mt = 0; mt < 4; ++mt) {
        for (int nt = 0; nt < 4; ++nt) {
            int ng = n0 + waveN * 64 + nt * 16 + lm;
            if (ng >= N) continue;
            #pragma unroll
            for (int v = 0; v < 8; ++v) {
                int mg = m0 + waveM * 64 + mt * 16 + v + 8 * lh;
                C[(size_t)mg * N + ng] = acc[mt][nt][v];
            }
        }
    }
}

// ---------------------------------------------------------------------------
// depthwise causal conv(4) + SiLU + softplus(dt) -> x_ssm, A_bar
// ---------------------------------------------------------------------------
__global__ void conv_dt_kernel(const float* __restrict__ proj,
                               const float* __restrict__ conv_w,
                               const float* __restrict__ conv_b,
                               const float* __restrict__ A_log,
                               const float* __restrict__ dt_bias,
                               float* __restrict__ xssm,
                               float* __restrict__ abar) {
    int idx = blockIdx.x * 256 + threadIdx.x;          // over BT*D_INNER
    if (idx >= BT * D_INNER) return;
    int d   = idx & (D_INNER - 1);
    int row = idx >> 11;
    int t   = row & 1023;

    float s = conv_b[d];
    #pragma unroll
    for (int k = 0; k < 4; ++k) {
        int tt = t - 3 + k;
        if (tt >= 0)
            s += proj[(size_t)(row + tt - t) * IN_FEAT + D_INNER + d] * conv_w[d * 4 + k];
    }
    float xc = s / (1.f + __expf(-s));                 // SiLU

    int head  = d >> 6;
    float dtl = proj[(size_t)row * IN_FEAT + 2 * D_INNER + 2 * D_STATE + head] + dt_bias[head];
    float dt  = (dtl > 20.f) ? dtl : log1pf(__expf(dtl));   // softplus
    xssm[(size_t)row * D_INNER + d] = dt * xc;
    if ((d & 63) == 0) {
        float ab = __expf(-dt * __expf(A_log[head]));
        abar[(size_t)row * N_HEADS + head] = fmaxf(ab, 1e-7f);
    }
}

// ---------------------------------------------------------------------------
// selective scan: 1 WG per (b, head); thread = (d in 0..63, s-block in 0..3)
// 16 states per thread in registers; cross-lane reduce over s-blocks.
// ---------------------------------------------------------------------------
__global__ __launch_bounds__(256)
void scan_kernel(const float* __restrict__ proj,
                 const float* __restrict__ xssm,
                 const float* __restrict__ abar,
                 const float* __restrict__ D_param,
                 float* __restrict__ y) {
    int b    = blockIdx.x >> 5;
    int head = blockIdx.x & 31;
    int tid  = threadIdx.x;
    int dloc = tid >> 2;
    int sblk = tid & 3;
    int d    = head * 64 + dloc;
    float Dp = D_param[d];

    float h[16];
    #pragma unroll
    for (int i = 0; i < 16; ++i) h[i] = 0.f;

    for (int t = 0; t < 1024; ++t) {
        int row   = b * 1024 + t;
        float ab  = abar[row * N_HEADS + head];                 // WG-uniform -> s_load
        float xdt = xssm[(size_t)row * D_INNER + d];
        const float* Bp = proj + (size_t)row * IN_FEAT + 2 * D_INNER + sblk * 16;
        const float* Cp = Bp + D_STATE;
        float yacc = 0.f;
        #pragma unroll
        for (int i = 0; i < 16; i += 4) {
            float4 Bv = *(const float4*)(Bp + i);
            float4 Cv = *(const float4*)(Cp + i);
            h[i]     = ab * h[i]     + xdt * Bv.x;  yacc += h[i]     * Cv.x;
            h[i + 1] = ab * h[i + 1] + xdt * Bv.y;  yacc += h[i + 1] * Cv.y;
            h[i + 2] = ab * h[i + 2] + xdt * Bv.z;  yacc += h[i + 2] * Cv.z;
            h[i + 3] = ab * h[i + 3] + xdt * Bv.w;  yacc += h[i + 3] * Cv.w;
        }
        yacc += __shfl_xor(yacc, 1);
        yacc += __shfl_xor(yacc, 2);
        if (sblk == 0)
            y[(size_t)row * D_INNER + d] = yacc + Dp * xdt;
    }
}

// ---------------------------------------------------------------------------
// gating + RMSNorm -> bf16 for final GEMM; 1 WG per token row
// ---------------------------------------------------------------------------
__global__ __launch_bounds__(256)
void gate_norm_kernel(const float* __restrict__ proj,
                      const float* __restrict__ y,
                      const float* __restrict__ norm_w,
                      __bf16* __restrict__ ynbf) {
    int row = blockIdx.x;
    int tid = threadIdx.x;
    float g[8];
    float ss = 0.f;
    #pragma unroll
    for (int i = 0; i < 8; ++i) {
        int c    = tid + i * 256;
        float z  = proj[(size_t)row * IN_FEAT + c];
        float yv = y[(size_t)row * D_INNER + c];
        float gv = yv * (z / (1.f + __expf(-z)));
        g[i] = gv;
        ss  += gv * gv;
    }
    #pragma unroll
    for (int off = 16; off > 0; off >>= 1) ss += __shfl_xor(ss, off);
    __shared__ float red[8];
    if ((tid & 31) == 0) red[tid >> 5] = ss;
    __syncthreads();
    float tot = 0.f;
    #pragma unroll
    for (int w = 0; w < 8; ++w) tot += red[w];
    float rms = rsqrtf(tot / (float)D_INNER + 1e-6f);
    #pragma unroll
    for (int i = 0; i < 8; ++i) {
        int c = tid + i * 256;
        ynbf[(size_t)row * D_INNER + c] = (__bf16)(g[i] * rms * norm_w[c]);
    }
}

// ---------------------------------------------------------------------------
extern "C" void kernel_launch(void* const* d_in, const int* in_sizes, int n_in,
                              void* d_out, int out_size, void* d_ws, size_t ws_size,
                              hipStream_t stream) {
    const float* x       = (const float*)d_in[0];
    const float* W_in    = (const float*)d_in[1];
    const float* W_out   = (const float*)d_in[2];
    const float* conv_w  = (const float*)d_in[3];
    const float* conv_b  = (const float*)d_in[4];
    const float* A_log   = (const float*)d_in[5];
    const float* D_param = (const float*)d_in[6];
    const float* dt_bias = (const float*)d_in[7];
    const float* norm_w  = (const float*)d_in[8];
    float* out = (float*)d_out;

    char* ws = (char*)d_ws;
    __bf16* xbf  = (__bf16*)(ws + OFF_XBF);
    __bf16* wibf = (__bf16*)(ws + OFF_WIBF);
    __bf16* wobf = (__bf16*)(ws + OFF_WOBF);
    float*  proj = (float*)(ws + OFF_PROJ);
    float*  xssm = (float*)(ws + OFF_XSSM);
    float*  abar = (float*)(ws + OFF_ABAR);
    float*  yraw = (float*)(ws + OFF_Y);
    __bf16* ynbf = (__bf16*)(ws + OFF_YNBF);

    // 1) bf16 casts for WMMA operands
    cast_f32_bf16<<<(BT * DIM + 255) / 256, 256, 0, stream>>>(x, xbf, BT * DIM);
    cast_f32_bf16<<<(IN_FEAT * DIM + 255) / 256, 256, 0, stream>>>(W_in, wibf, IN_FEAT * DIM);
    cast_f32_bf16<<<(DIM * D_INNER + 255) / 256, 256, 0, stream>>>(W_out, wobf, DIM * D_INNER);

    // 2) proj = x @ W_in^T  (M=2048, N=4256, K=1024)
    gemm_bf16_wmma<<<dim3((IN_FEAT + BN - 1) / BN, BT / BM), 256, 0, stream>>>(
        xbf, wibf, proj, BT, IN_FEAT, DIM);

    // 3) conv + SiLU + dt/softplus -> x_ssm, A_bar
    conv_dt_kernel<<<(BT * D_INNER + 255) / 256, 256, 0, stream>>>(
        proj, conv_w, conv_b, A_log, dt_bias, xssm, abar);

    // 4) selective scan -> y (includes D*x_ssm skip)
    scan_kernel<<<64, 256, 0, stream>>>(proj, xssm, abar, D_param, yraw);

    // 5) gate + RMSNorm -> bf16
    gate_norm_kernel<<<BT, 256, 0, stream>>>(proj, yraw, norm_w, ynbf);

    // 6) out = yn @ W_out^T  (M=2048, N=1024, K=2048)
    gemm_bf16_wmma<<<dim3(DIM / BN, BT / BM), 256, 0, stream>>>(
        ynbf, wobf, out, BT, DIM, D_INNER);
}
